// TemporalPoseTransformer_57956288692768
// MI455X (gfx1250) — matrix operand order repaired and checked by reading, not compiled, not run on hardware
//
#include <hip/hip_runtime.h>
#include <cstdint>
#include <cstddef>

typedef __attribute__((ext_vector_type(16))) __bf16    v16bf;
typedef __attribute__((ext_vector_type(8)))  float     v8f;
typedef __attribute__((ext_vector_type(4)))  uint32_t  v4u;
typedef __attribute__((ext_vector_type(2)))  float     v2f;
typedef __attribute__((ext_vector_type(4)))  int       v4i;
typedef __attribute__((ext_vector_type(8)))  int       v8i;

union BF16Frag { v16bf v; uint32_t u[8]; v4u q[2]; };

__device__ __forceinline__ unsigned short f2bf(float f) {
    unsigned int u = __float_as_uint(f);
    unsigned int r = u + 0x7FFFu + ((u >> 16) & 1u);   // round-to-nearest-even
    return (unsigned short)(r >> 16);
}
__device__ __forceinline__ uint32_t pack2bf(float lo, float hi) {
    return (uint32_t)f2bf(lo) | ((uint32_t)f2bf(hi) << 16);
}

#if defined(__HIP_DEVICE_COMPILE__) && __has_builtin(__builtin_amdgcn_tensor_load_to_lds)
#define USE_TDM 1
#else
#define USE_TDM 0
#endif

#if USE_TDM
// TDM load of a [tile_rows x 32] bf16 tile (row stride K elements) into LDS
// with 4-dword padding every 16 dwords (=> 20-dword LDS row stride).
__device__ __forceinline__ void tdm_load_tile(uint32_t lds_byte_addr,
                                              const void* gaddr,
                                              int K, int rows_total,
                                              int tile_rows) {
    uint64_t ga = (uint64_t)(uintptr_t)gaddr;
    v4u g0;
    g0[0] = 1u;                                            // count=1, user mode
    g0[1] = lds_byte_addr;                                 // lds_addr
    g0[2] = (uint32_t)(ga & 0xFFFFFFFFu);                  // global_addr[31:0]
    g0[3] = (uint32_t)((ga >> 32) & 0x01FFFFFFu) | (2u << 30); // [56:32] | type=2
    v8i g1;
    g1[0] = (int)((1u << 16) |                             // data_size = 2 bytes
                  (1u << 20) |                             // pad_enable
                  (3u << 22) |                             // pad_interval: 16 dwords
                  (3u << 25));                             // pad_amount: 4 dwords
    g1[1] = (int)(((uint32_t)K & 0xFFFFu) << 16);          // tensor_dim0[15:0]
    g1[2] = (int)((((uint32_t)K >> 16) & 0xFFFFu) |
                  (((uint32_t)rows_total & 0xFFFFu) << 16)); // dim0 hi | dim1 lo
    g1[3] = (int)((((uint32_t)rows_total >> 16) & 0xFFFFu) |
                  (32u << 16));                            // dim1 hi | tile_dim0=32
    g1[4] = tile_rows;                                     // tile_dim1 (tile_dim2=0)
    g1[5] = K;                                             // tensor_dim0_stride
    g1[6] = 0;
    g1[7] = 0;
    v4i z4 = (v4i){0, 0, 0, 0};
#if __clang_major__ >= 23
    v8i z8 = (v8i){0, 0, 0, 0, 0, 0, 0, 0};
    __builtin_amdgcn_tensor_load_to_lds(g0, g1, z4, z4, z8, 0);
#else
    __builtin_amdgcn_tensor_load_to_lds(g0, g1, z4, z4, 0);
#endif
}
#endif

// ---------------------------------------------------------------------------
// f32 -> bf16 bulk convert (one u32 pair per thread)
// ---------------------------------------------------------------------------
__global__ void cvt_bf16_kernel(const float* __restrict__ s,
                                unsigned short* __restrict__ d) {
    size_t i = (size_t)blockIdx.x * 256 + threadIdx.x;
    v2f f = *reinterpret_cast<const v2f*>(s + 2 * i);
    reinterpret_cast<uint32_t*>(d)[i] = pack2bf(f.x, f.y);
}

// ---------------------------------------------------------------------------
// x = tokens + sinusoidal PE
// ---------------------------------------------------------------------------
__global__ void pe_add_kernel(const float* __restrict__ tokens,
                              float* __restrict__ x) {
    int i = blockIdx.x * 256 + threadIdx.x;
    int d = i & 255;
    int t = (i >> 8) & 2047;
    int j = d >> 1;
    float freq = __expf(-(float)(2 * j) * (9.210340371976184f / 256.0f));
    float ang  = (float)t * freq;
    float pe   = (d & 1) ? __cosf(ang) : __sinf(ang);
    x[i] = tokens[i] + pe;
}

// ---------------------------------------------------------------------------
// LayerNorm, one wave per 256-wide row; writes f32 (yf) or bf16 (yb).
// ---------------------------------------------------------------------------
__global__ void ln_kernel(const float* __restrict__ x,
                          const float* __restrict__ g,
                          const float* __restrict__ b,
                          float* __restrict__ yf,
                          unsigned short* __restrict__ yb) {
    const int D = 256;
    int lane = threadIdx.x & 31;
    int row  = blockIdx.x * 8 + (threadIdx.x >> 5);
    const float* xr = x + (size_t)row * D;
    float v[8];
    float s = 0.f;
#pragma unroll
    for (int i = 0; i < 8; ++i) { v[i] = xr[lane + 32 * i]; s += v[i]; }
#pragma unroll
    for (int off = 16; off; off >>= 1) s += __shfl_xor(s, off, 32);
    float mu = s * (1.0f / 256.0f);
    float q = 0.f;
#pragma unroll
    for (int i = 0; i < 8; ++i) { float dv = v[i] - mu; q += dv * dv; }
#pragma unroll
    for (int off = 16; off; off >>= 1) q += __shfl_xor(q, off, 32);
    float rs = rsqrtf(q * (1.0f / 256.0f) + 1e-5f);
#pragma unroll
    for (int i = 0; i < 8; ++i) {
        int d = lane + 32 * i;
        float val = (v[i] - mu) * rs * g[d] + b[d];
        if (yb) yb[(size_t)row * D + d] = f2bf(val);
        else    yf[(size_t)row * D + d] = val;
    }
}

// ---------------------------------------------------------------------------
// WMMA GEMM on pre-converted bf16:  C = epilogue( A[M,K] @ W[N,K]^T + bias )
// flags bit0: exact GELU, bit1: += R (f32), bit2: store bf16 to Cb else f32 C
// Block 128x64, 8 waves. Double-buffered LDS; wave 0 software-pipelines TDM
// tensor_load_to_lds (tile i+1 in flight while tile i computes), releasing
// data with s_wait_tensorcnt + workgroup split barrier. Fallback: b128 copies.
// ---------------------------------------------------------------------------
__global__ __launch_bounds__(256)
void gemm_bf16_wmma_kernel(const unsigned short* __restrict__ A,
                           const unsigned short* __restrict__ W,
                           const float* __restrict__ bias,
                           const float* __restrict__ R,
                           float* __restrict__ C,
                           unsigned short* __restrict__ Cb,
                           int M, int N, int K, int flags) {
    __shared__ __align__(16) uint32_t lA[2][128 * 20];
    __shared__ __align__(16) uint32_t lB[2][64 * 20];

    const int tid    = threadIdx.x;
    const int lane   = tid & 31;
    const int w      = tid >> 5;
    const int lane15 = lane & 15;
    const int hi     = (lane >> 4) & 1;
    const int m0     = blockIdx.x * 128;
    const int n0     = blockIdx.y * 64;
    const int nk     = K >> 5;                 // K tiles of 32

    v8f acc[4];
#pragma unroll
    for (int j = 0; j < 4; ++j) acc[j] = (v8f){0.f,0.f,0.f,0.f,0.f,0.f,0.f,0.f};

#if USE_TDM
    const int wid = __builtin_amdgcn_readfirstlane(w);     // scalar wave id
    const uint32_t ldsA0 = (uint32_t)(uintptr_t)&lA[0][0];
    const uint32_t ldsA1 = (uint32_t)(uintptr_t)&lA[1][0];
    const uint32_t ldsB0 = (uint32_t)(uintptr_t)&lB[0][0];
    const uint32_t ldsB1 = (uint32_t)(uintptr_t)&lB[1][0];
    if (wid == 0) {                            // prologue: tile 0 in flight
        tdm_load_tile(ldsA0, A + (size_t)m0 * K, K, M - m0, 128);
        tdm_load_tile(ldsB0, W + (size_t)n0 * K, K, N - n0, 64);
    }
#endif

    for (int i = 0; i < nk; ++i) {
        const int buf = i & 1;
#if USE_TDM
        if (wid == 0) {
            if (i + 1 < nk) {                  // pipeline next tile
                int k1 = (i + 1) * 32;
                tdm_load_tile(buf ? ldsA0 : ldsA1, A + (size_t)m0 * K + k1, K, M - m0, 128);
                tdm_load_tile(buf ? ldsB0 : ldsB1, W + (size_t)n0 * K + k1, K, N - n0, 64);
                __builtin_amdgcn_s_wait_tensorcnt(2);  // tile i done, i+1 in flight
            } else {
                __builtin_amdgcn_s_wait_tensorcnt(0);
            }
        }
        __syncthreads();
        asm volatile("" ::: "memory");         // LDS written behind compiler's back
#else
        {
            const int k0 = i * 32;
#pragma unroll
            for (int ii = 0; ii < 2; ++ii) {   // A: 512 v4u, 2/thread
                int idx = ii * 256 + tid;
                int r = idx >> 2, p4 = idx & 3;
                v4u t4 = reinterpret_cast<const v4u*>(A + (size_t)(m0 + r) * K + k0)[p4];
                *reinterpret_cast<v4u*>(lA[buf] + r * 20 + p4 * 4) = t4;
            }
            int n = tid >> 2, p4 = tid & 3;    // B: 256 v4u, 1/thread
            v4u t4 = reinterpret_cast<const v4u*>(W + (size_t)(n0 + n) * K + k0)[p4];
            *reinterpret_cast<v4u*>(lB[buf] + n * 20 + p4 * 4) = t4;
        }
        __syncthreads();
#endif
        const uint32_t* aRow = lA[buf] + (w * 16 + lane15) * 20;
        BF16Frag a;                            // two ds_load_b128
        a.q[0] = *reinterpret_cast<const v4u*>(aRow + hi * 4);
        a.q[1] = *reinterpret_cast<const v4u*>(aRow + 8 + hi * 4);
#pragma unroll
        for (int j = 0; j < 4; ++j) {
            const uint32_t* bRow = lB[buf] + (j * 16 + lane15) * 20 + hi * 4;
            BF16Frag bf;
            bf.q[0] = *reinterpret_cast<const v4u*>(bRow);
            bf.q[1] = *reinterpret_cast<const v4u*>(bRow + 8);
            acc[j] = __builtin_amdgcn_wmma_f32_16x16x32_bf16(
                false, a.v, false, bf.v, (short)0, acc[j], false, false);
        }
        __syncthreads();                       // frees buf for TDM overwrite at i+2
    }

    const int mbase = m0 + w * 16 + hi * 8;
    float rv[4][8];
    if (flags & 2) {                           // batch residual loads up front
#pragma unroll
        for (int j = 0; j < 4; ++j) {
            int gn = n0 + j * 16 + lane15;
#pragma unroll
            for (int v = 0; v < 8; ++v)
                rv[j][v] = R[(size_t)(mbase + v) * N + gn];
        }
    }
#pragma unroll
    for (int j = 0; j < 4; ++j) {
        int gn = n0 + j * 16 + lane15;
        float bv = bias[gn];
#pragma unroll
        for (int v = 0; v < 8; ++v) {
            int gm = mbase + v;
            float val = acc[j][v] + bv;
            if (flags & 1) val = 0.5f * val * (1.0f + erff(val * 0.70710678118f));
            if (flags & 2) val += rv[j][v];
            if (flags & 4) Cb[(size_t)gm * N + gn] = f2bf(val);
            else           C [(size_t)gm * N + gn] = val;
        }
    }
}

// ---------------------------------------------------------------------------
// Sliding-window attention (W=64) via WMMA, all-bf16 I/O.
// One wave per (b, h, 16-query tile); 4 waves/block.
// qkvb: [B*T, 768] bf16; q|k|v cols 0|256|512, head offset h*64. T=2048.
// Scale 1/sqrt(64) applied to f32 scores (staging stays a pure copy).
// ---------------------------------------------------------------------------
__global__ __launch_bounds__(128)
void attn_wmma_kernel(const unsigned short* __restrict__ qkvb,
                      unsigned short* __restrict__ obb) {
    __shared__ __align__(16) uint32_t       lQ[4][16 * 40];
    __shared__ __align__(16) uint32_t       lK[4][16 * 40];
    __shared__ __align__(16) unsigned short lP[4][16 * 96];
    __shared__ __align__(16) uint32_t       lV[4][64 * 20];

    const int lane   = threadIdx.x & 31;
    const int w      = threadIdx.x >> 5;
    const int lane15 = lane & 15;
    const int hi     = (lane >> 4) & 1;
    const int g      = blockIdx.x * 4 + w;
    const int h      = g & 3;
    const int qt0    = ((g >> 2) & 127) * 16;
    const int b      = g >> 9;
    const int kb     = qt0 - 64;

    uint32_t*       q_ = lQ[w];
    uint32_t*       k_ = lK[w];
    unsigned short* p_ = lP[w];
    uint32_t*       v_ = lV[w];
    const size_t base = (size_t)b * 2048 * 768;

    // stage Q: one b128 global load + one b128 LDS store per lane
    {
        int r = lane >> 1, part = lane & 1;
        v4u t4 = reinterpret_cast<const v4u*>(qkvb + base + (size_t)(qt0 + r) * 768 + h * 64)[part];
        *reinterpret_cast<v4u*>(q_ + r * 40 + part * 4) = t4;
    }
    if (!hi) {
#pragma unroll
        for (int i = 0; i < 16; ++i) p_[i * 96 + 80 + lane15] = 0;
    }
    __syncthreads();

    // ---- scores: 5 key tiles x 2 K-chunks
    v8f sc[5];
#pragma unroll
    for (int t = 0; t < 5; ++t) {
        {
            int r = lane >> 1, part = lane & 1;
            int j  = kb + t * 16 + r;
            int jc = j < 0 ? 0 : j;
            v4u t4 = reinterpret_cast<const v4u*>(qkvb + base + (size_t)jc * 768 + 256 + h * 64)[part];
            *reinterpret_cast<v4u*>(k_ + r * 40 + part * 4) = t4;
        }
        __syncthreads();
        v8f acc = (v8f){0.f,0.f,0.f,0.f,0.f,0.f,0.f,0.f};
#pragma unroll
        for (int c = 0; c < 2; ++c) {
            BF16Frag af, bf;
            const uint32_t* ar = q_ + lane15 * 40 + c * 16;
            af.q[0] = *reinterpret_cast<const v4u*>(ar + hi * 4);
            af.q[1] = *reinterpret_cast<const v4u*>(ar + 8 + hi * 4);
            const uint32_t* br = k_ + lane15 * 40 + c * 16 + hi * 4;
            bf.q[0] = *reinterpret_cast<const v4u*>(br);
            bf.q[1] = *reinterpret_cast<const v4u*>(br + 8);
            acc = __builtin_amdgcn_wmma_f32_16x16x32_bf16(
                false, af.v, false, bf.v, (short)0, acc, false, false);
        }
        sc[t] = acc;
        __syncthreads();
    }

    // ---- mask + softmax (row = hi*8 + v, key col = lane15), scale 0.125
#pragma unroll
    for (int v = 0; v < 8; ++v) {
        const int irow = qt0 + hi * 8 + v;
        float mx = -1e30f;
#pragma unroll
        for (int t = 0; t < 5; ++t) {
            int j = kb + t * 16 + lane15;
            bool ok = (j >= 0) && (j <= irow) && (j >= irow - 63);
            float s = ok ? sc[t][v] * 0.125f : -1e30f;
            sc[t][v] = s;
            mx = fmaxf(mx, s);
        }
#pragma unroll
        for (int off = 1; off < 16; off <<= 1) mx = fmaxf(mx, __shfl_xor(mx, off, 32));
        float sum = 0.f;
#pragma unroll
        for (int t = 0; t < 5; ++t) { float e = __expf(sc[t][v] - mx); sc[t][v] = e; sum += e; }
#pragma unroll
        for (int off = 1; off < 16; off <<= 1) sum += __shfl_xor(sum, off, 32);
        float is = 1.0f / sum;
#pragma unroll
        for (int t = 0; t < 5; ++t)
            p_[(hi * 8 + v) * 96 + t * 16 + lane15] = f2bf(sc[t][v] * is);
    }
    __syncthreads();

    // ---- P @ V: 3 chunks of 32 keys; V transposed (raw bf16 bit packing)
    v8f oa[4];
#pragma unroll
    for (int j4 = 0; j4 < 4; ++j4) oa[j4] = (v8f){0.f,0.f,0.f,0.f,0.f,0.f,0.f,0.f};

#pragma unroll
    for (int c = 0; c < 3; ++c) {
#pragma unroll
        for (int i = 0; i < 32; ++i) {
            int idx = i * 32 + lane;
            int n = idx >> 4, qq = idx & 15;
            int j0 = kb + c * 32 + 2 * qq;
            int j1 = j0 + 1;
            int jc0 = j0 < 0 ? 0 : (j0 > 2047 ? 2047 : j0);
            int jc1 = j1 < 0 ? 0 : (j1 > 2047 ? 2047 : j1);
            uint32_t lo  = qkvb[base + (size_t)jc0 * 768 + 512 + h * 64 + n];
            uint32_t hi2 = qkvb[base + (size_t)jc1 * 768 + 512 + h * 64 + n];
            v_[n * 20 + qq] = lo | (hi2 << 16);
        }
        __syncthreads();
        BF16Frag af;
        const uint32_t* ap = reinterpret_cast<const uint32_t*>(p_) + lane15 * 48 + c * 16;
        af.q[0] = *reinterpret_cast<const v4u*>(ap + hi * 4);
        af.q[1] = *reinterpret_cast<const v4u*>(ap + 8 + hi * 4);
#pragma unroll
        for (int j4 = 0; j4 < 4; ++j4) {
            const uint32_t* br = v_ + (j4 * 16 + lane15) * 20 + hi * 4;
            BF16Frag bf;
            bf.q[0] = *reinterpret_cast<const v4u*>(br);
            bf.q[1] = *reinterpret_cast<const v4u*>(br + 8);
            oa[j4] = __builtin_amdgcn_wmma_f32_16x16x32_bf16(
                false, af.v, false, bf.v, (short)0, oa[j4], false, false);
        }
        __syncthreads();
    }

#pragma unroll
    for (int j4 = 0; j4 < 4; ++j4) {
#pragma unroll
        for (int v = 0; v < 8; ++v) {
            int gm = qt0 + hi * 8 + v;
            obb[(size_t)(b * 2048 + gm) * 256 + h * 64 + j4 * 16 + lane15] = f2bf(oa[j4][v]);
        }
    }
}

// ---------------------------------------------------------------------------
extern "C" void kernel_launch(void* const* d_in, const int* in_sizes, int n_in,
                              void* d_out, int out_size, void* d_ws, size_t ws_size,
                              hipStream_t stream) {
    const float* tokens = (const float*)d_in[0];
    const float* Wqkv   = (const float*)d_in[1];
    const float* bqkv   = (const float*)d_in[2];
    const float* Wo     = (const float*)d_in[3];
    const float* bo     = (const float*)d_in[4];
    const float* W1     = (const float*)d_in[5];
    const float* b1     = (const float*)d_in[6];
    const float* W2     = (const float*)d_in[7];
    const float* b2     = (const float*)d_in[8];
    const float* ln1_g  = (const float*)d_in[9];
    const float* ln1_b  = (const float*)d_in[10];
    const float* ln2_g  = (const float*)d_in[11];
    const float* ln2_b  = (const float*)d_in[12];
    const float* lnf_g  = (const float*)d_in[13];
    const float* lnf_b  = (const float*)d_in[14];

    const int Bsz = 4, T = 2048, D = 256, F = 1024, Lnum = 4;
    const int M = Bsz * T;                               // 8192

    float*          x     = (float*)d_ws;                               // M*D f32
    unsigned short* x2b   = (unsigned short*)(x + (size_t)M * D);       // M*D
    unsigned short* qkvb  = x2b  + (size_t)M * D;                       // M*3D
    unsigned short* obb   = qkvb + (size_t)M * 3 * D;                   // M*D
    unsigned short* hb    = obb  + (size_t)M * D;                       // M*F
    unsigned short* wqkvb = hb    + (size_t)M * F;                      // L*3D*D
    unsigned short* wob   = wqkvb + (size_t)Lnum * 3 * D * D;           // L*D*D
    unsigned short* w1b   = wob   + (size_t)Lnum * D * D;               // L*F*D
    unsigned short* w2b   = w1b   + (size_t)Lnum * F * D;               // L*D*F

    // ---- one-time weight conversion f32 -> bf16
    cvt_bf16_kernel<<<(Lnum * 3 * D * D) / 512, 256, 0, stream>>>(Wqkv, wqkvb);
    cvt_bf16_kernel<<<(Lnum * D * D)     / 512, 256, 0, stream>>>(Wo,   wob);
    cvt_bf16_kernel<<<(Lnum * F * D)     / 512, 256, 0, stream>>>(W1,   w1b);
    cvt_bf16_kernel<<<(Lnum * D * F)     / 512, 256, 0, stream>>>(W2,   w2b);

    pe_add_kernel<<<(M * D) / 256, 256, 0, stream>>>(tokens, x);

    for (int l = 0; l < Lnum; ++l) {
        const unsigned short* Wqkv_l = wqkvb + (size_t)l * 3 * D * D;
        const unsigned short* Wo_l   = wob   + (size_t)l * D * D;
        const unsigned short* W1_l   = w1b   + (size_t)l * F * D;
        const unsigned short* W2_l   = w2b   + (size_t)l * D * F;
        const float* bqkv_l = bqkv + (size_t)l * 3 * D;
        const float* bo_l   = bo   + (size_t)l * D;
        const float* b1_l   = b1   + (size_t)l * F;
        const float* b2_l   = b2   + (size_t)l * D;

        ln_kernel<<<M / 8, 256, 0, stream>>>(x, ln1_g + l * D, ln1_b + l * D,
                                             nullptr, x2b);

        gemm_bf16_wmma_kernel<<<dim3(M / 128, (3 * D) / 64), 256, 0, stream>>>(
            x2b, Wqkv_l, bqkv_l, nullptr, nullptr, qkvb, M, 3 * D, D, 4);

        attn_wmma_kernel<<<(Bsz * 4 * (T / 16)) / 4, 128, 0, stream>>>(qkvb, obb);

        gemm_bf16_wmma_kernel<<<dim3(M / 128, D / 64), 256, 0, stream>>>(
            obb, Wo_l, bo_l, x, x, nullptr, M, D, D, 2);       // x += o@Wo+bo

        ln_kernel<<<M / 8, 256, 0, stream>>>(x, ln2_g + l * D, ln2_b + l * D,
                                             nullptr, x2b);

        gemm_bf16_wmma_kernel<<<dim3(M / 128, F / 64), 256, 0, stream>>>(
            x2b, W1_l, b1_l, nullptr, nullptr, hb, M, F, D, 1 | 4);  // gelu->bf16

        gemm_bf16_wmma_kernel<<<dim3(M / 128, D / 64), 256, 0, stream>>>(
            hb, W2_l, b2_l, x, x, nullptr, M, D, F, 2);        // x += h@W2+b2
    }

    ln_kernel<<<M / 8, 256, 0, stream>>>(x, lnf_g, lnf_b, (float*)d_out, nullptr);
}
